// ScaledDotProductAttention_901943132790
// MI455X (gfx1250) — compile-verified, matrix operand order checked by reading, not run on hardware
//
#include <hip/hip_runtime.h>
#include <hip/hip_bf16.h>

// Flash-attention forward, CDNA5 (gfx1250), wave32, bf16 WMMA with fp32 accum.
// B=2, H=16, S=2048, D=64, causal. Mask input ignored (causality is analytic).
// Round 4: permuted-k P/V LDS layout (packed b64/b32 stores), b128 K staging
//          stores, exp2-domain softmax, rcp epilogue.

typedef __attribute__((ext_vector_type(16))) __bf16 v16bf;
typedef __attribute__((ext_vector_type(8)))  __bf16 v8bf;
typedef __attribute__((ext_vector_type(4)))  __bf16 v4bf;
typedef __attribute__((ext_vector_type(8)))  float  v8f;
typedef __attribute__((ext_vector_type(4)))  float  f32x4;

#define S_LEN 2048
#define D_DIM 64
#define WAVES 4            // waves per workgroup
#define ROWS_PER_WAVE 16   // Q rows per wave (one WMMA M-tile)
#define KEY_BLOCK 64       // keys per workgroup iteration
#define LDSPITCH 72        // padded row pitch (bf16 elems): 144B -> conflict-free
// Q pre-scale: 1/sqrt(64) folded with log2(e) for exp2-domain softmax.
#define QSCALE (0.125f * 1.442695040888963f)

// ---------------- intra-16-lane butterfly reductions (VALU) ----------------
#if __has_builtin(__builtin_amdgcn_permlane16)
__device__ __forceinline__ float permx16(float x, unsigned s1, unsigned s2) {
  int xi = __builtin_bit_cast(int, x);
  int r  = __builtin_amdgcn_permlane16(xi, xi, s1, s2, false, false);
  return __builtin_bit_cast(float, r);
}
__device__ __forceinline__ float row_max16(float x) {
  x = fmaxf(x, permx16(x, 0x67452301u, 0xEFCDAB89u));  // xor 1
  x = fmaxf(x, permx16(x, 0x54761032u, 0xDCFE98BAu));  // xor 2
  x = fmaxf(x, permx16(x, 0x32107654u, 0xBA98FEDCu));  // xor 4
  x = fmaxf(x, permx16(x, 0xFEDCBA98u, 0x76543210u));  // xor 8
  return x;
}
__device__ __forceinline__ float row_sum16(float x) {
  x += permx16(x, 0x67452301u, 0xEFCDAB89u);
  x += permx16(x, 0x54761032u, 0xDCFE98BAu);
  x += permx16(x, 0x32107654u, 0xBA98FEDCu);
  x += permx16(x, 0xFEDCBA98u, 0x76543210u);
  return x;
}
#else
__device__ __forceinline__ float row_max16(float x) {
  for (int m = 1; m <= 8; m <<= 1) x = fmaxf(x, __shfl_xor(x, m, 32));
  return x;
}
__device__ __forceinline__ float row_sum16(float x) {
  for (int m = 1; m <= 8; m <<= 1) x += __shfl_xor(x, m, 32);
  return x;
}
#endif

#if __has_builtin(__builtin_amdgcn_exp2f)
__device__ __forceinline__ float fast_exp2(float x) { return __builtin_amdgcn_exp2f(x); }
#else
__device__ __forceinline__ float fast_exp2(float x) { return exp2f(x); }
#endif
#if __has_builtin(__builtin_amdgcn_rcpf)
__device__ __forceinline__ float fast_rcp(float x) { return __builtin_amdgcn_rcpf(x); }
#else
__device__ __forceinline__ float fast_rcp(float x) { return 1.0f / x; }
#endif

// Assemble a 16x32 bf16 fragment from a padded row-major bf16 LDS row.
// Halves 0..7 -> k = off+0..7, halves 8..15 -> k = off+16..23 (ISA A/B layout,
// per-half k-base folded into `off`). Two aligned 16B runs -> ds_load_b128.
__device__ __forceinline__ v16bf frag_from_lds(const __bf16* row, int off) {
  v8bf lo = *(const v8bf*)(row + off);
  v8bf hi = *(const v8bf*)(row + off + 16);
  return __builtin_shufflevector(lo, hi, 0, 1, 2, 3, 4, 5, 6, 7,
                                 8, 9, 10, 11, 12, 13, 14, 15);
}

// Same fragment pattern straight from a row-major fp32 matrix (Q only).
__device__ __forceinline__ v16bf load_frag_global(const float* __restrict__ p,
                                                  float scale) {
  v16bf f;
#pragma unroll
  for (int i = 0; i < 8; ++i) f[i]     = (__bf16)(p[i]      * scale);
#pragma unroll
  for (int i = 0; i < 8; ++i) f[8 + i] = (__bf16)(p[16 + i] * scale);
  return f;
}

__global__ __launch_bounds__(WAVES * 32, 2)
void fa_fwd_causal_bf16wmma(const float* __restrict__ Q,
                            const float* __restrict__ K,
                            const float* __restrict__ V,
                            float* __restrict__ O) {
  const int tid  = threadIdx.x;
  const int lane = tid & 31;
  const int wave = tid >> 5;
  const int tiles_per_bh = S_LEN / (WAVES * ROWS_PER_WAVE);
  const int bh      = blockIdx.x / tiles_per_bh;
  const int tile    = blockIdx.x % tiles_per_bh;
  const int wg_base = tile * (WAVES * ROWS_PER_WAVE);
  const int m_base  = wg_base + wave * ROWS_PER_WAVE;

  const size_t base = (size_t)bh * S_LEN * D_DIM;
  const float* q = Q + base;
  const float* k = K + base;
  const float* v = V + base;
  float*       o = O + base;

  const int col  = lane & 15;   // column within a 16-wide N tile (C/B layout)
  const int half = lane >> 4;   // which 16-lane half of the wave
  const int kb   = half * 8;    // per-half k-base inside a 32-wide K chunk

  // ---- Q tile (16x64) as two bf16 A-fragments, pre-scaled --------------
  const float* qrow = q + (size_t)(m_base + col) * D_DIM + kb;
  const v16bf aq0 = load_frag_global(qrow,      QSCALE);
  const v16bf aq1 = load_frag_global(qrow + 32, QSCALE);

  v8f acc[4]; // O accumulators, four 16-wide N tiles covering D=64
#pragma unroll
  for (int t = 0; t < 4; ++t) acc[t] = (v8f){};
  float mrow[8], lrow[8];
#pragma unroll
  for (int r = 0; r < 8; ++r) { mrow[r] = -1e30f; lrow[r] = 0.f; }

  // Cooperative bf16 staging. Vtld and plds use a PERMUTED key order
  // p = (key%16)*4 + key/16 (both sides agree, so P*V contracts correctly).
  __shared__ __align__(16) __bf16 Klds[KEY_BLOCK][LDSPITCH];   // [key][d]
  __shared__ __align__(16) __bf16 Vtld[D_DIM][LDSPITCH];       // [d][perm key]
  __shared__ __align__(16) __bf16 plds[WAVES][ROWS_PER_WAVE][LDSPITCH];

  // Uniform across the workgroup: causal needs key blocks 0..tile.
  const int nblocks = tile + 1;

  for (int j = 0; j < nblocks; ++j) {
    const int key_base = j * KEY_BLOCK;

    __syncthreads();  // previous iteration's LDS reads complete

    // ---- stage K block (contiguous 16KB): b128 loads -> b128 stores ----
    {
      const float* src = k + (size_t)key_base * D_DIM;
#pragma unroll
      for (int g = 0; g < 4; ++g) {
        const int e = g * 1024 + tid * 8;         // flat fp32 element index
        const f32x4 x0 = *(const f32x4*)(src + e);
        const f32x4 x1 = *(const f32x4*)(src + e + 4);
        v8bf y;
#pragma unroll
        for (int i = 0; i < 4; ++i) { y[i] = (__bf16)x0[i]; y[4 + i] = (__bf16)x1[i]; }
        *(v8bf*)&Klds[e >> 6][e & 63] = y;
      }
    }
    // ---- stage V transposed in permuted key order, packed u32 stores ----
    // Thread handles keys k0 = npair*16+c and k1 = k0+16, whose permuted
    // positions p0 = c*4+npair and p0+1 are adjacent in Vtld.
    {
      const int kp    = tid & 31;
      const int c     = kp >> 1;          // key column 0..15
      const int npair = (kp & 1) * 2;     // key tile pair 0 or 2
      const int p0    = c * 4 + npair;    // permuted position (even)
      const int db    = (tid >> 5) * 16;  // d range handled by this thread
      const float* r0 = v + (size_t)(key_base + npair * 16 + c) * D_DIM + db;
      const float* r1 = r0 + (size_t)16 * D_DIM;
      f32x4 a0[4], a1[4];
#pragma unroll
      for (int g = 0; g < 4; ++g) {
        a0[g] = *(const f32x4*)(r0 + 4 * g);
        a1[g] = *(const f32x4*)(r1 + 4 * g);
      }
#pragma unroll
      for (int d = 0; d < 16; ++d) {
        union { __bf16 h[2]; unsigned u; } pk;
        pk.h[0] = (__bf16)a0[d >> 2][d & 3];
        pk.h[1] = (__bf16)a1[d >> 2][d & 3];
        *(unsigned*)&Vtld[db + d][p0] = pk.u;
      }
    }

    __syncthreads();  // staging visible to all waves

    // ---- scores S = (Q*scale) * K^T : four 16x16 tiles, 8 bf16 WMMAs ----
    v8f s[4];
#pragma unroll
    for (int n = 0; n < 4; ++n) {
      const __bf16* krow = &Klds[n * 16 + col][0];
      const v16bf bk0 = frag_from_lds(krow, kb);
      const v16bf bk1 = frag_from_lds(krow, 32 + kb);
      v8f sn = (v8f){};
      sn = __builtin_amdgcn_wmma_f32_16x16x32_bf16(false, aq0, false, bk0,
                                                   (short)0, sn, false, false);
      sn = __builtin_amdgcn_wmma_f32_16x16x32_bf16(false, aq1, false, bk1,
                                                   (short)0, sn, false, false);
      s[n] = sn;
    }

    // ---- causal mask (only the diagonal block needs it) ----
    if (key_base + KEY_BLOCK - 1 > m_base) {
#pragma unroll
      for (int n = 0; n < 4; ++n)
#pragma unroll
        for (int r = 0; r < 8; ++r) {
          const int qg = m_base + half * 8 + r;          // global query row
          if (key_base + n * 16 + col > qg) s[n][r] = -1e30f;
        }
    }

    // ---- online softmax in exp2 domain (v_exp_f32 is natively 2^x) ----
    float corr[8];
#pragma unroll
    for (int r = 0; r < 8; ++r) {
      float bm = fmaxf(fmaxf(s[0][r], s[1][r]), fmaxf(s[2][r], s[3][r]));
      bm = row_max16(bm);
      const float mnew = fmaxf(mrow[r], bm);
      corr[r] = fast_exp2(mrow[r] - mnew);
      mrow[r] = mnew;
      float p0 = fast_exp2(s[0][r] - mnew);
      float p1 = fast_exp2(s[1][r] - mnew);
      float p2 = fast_exp2(s[2][r] - mnew);
      float p3 = fast_exp2(s[3][r] - mnew);
      const float ps = row_sum16((p0 + p1) + (p2 + p3));
      lrow[r] = lrow[r] * corr[r] + ps;
      s[0][r] = p0; s[1][r] = p1; s[2][r] = p2; s[3][r] = p3;  // reuse as P
    }
#pragma unroll
    for (int t = 0; t < 4; ++t)
#pragma unroll
      for (int r = 0; r < 8; ++r) acc[t][r] *= corr[r];

    // ---- P (C layout -> A layout) via LDS, permuted k: one b64 per row ----
#pragma unroll
    for (int r = 0; r < 8; ++r) {
      v4bf pv;
#pragma unroll
      for (int n = 0; n < 4; ++n) pv[n] = (__bf16)s[n][r];
      *(v4bf*)&plds[wave][half * 8 + r][col * 4] = pv;
    }

    const __bf16* prow = &plds[wave][col][0];
    const v16bf ap0 = frag_from_lds(prow, kb);
    const v16bf ap1 = frag_from_lds(prow, 32 + kb);

    // ---- O += P * V : 4 N-tiles x 2 K-chunks, fragments from Vt LDS ----
#pragma unroll
    for (int t = 0; t < 4; ++t) {
      const __bf16* vrow = &Vtld[t * 16 + col][0];
#pragma unroll
      for (int c = 0; c < 2; ++c) {
        const v16bf bv = frag_from_lds(vrow, c * 32 + kb);
        acc[t] = __builtin_amdgcn_wmma_f32_16x16x32_bf16(
            false, (c == 0 ? ap0 : ap1), false, bv, (short)0, acc[t],
            false, false);
      }
    }
  }

  // ---- epilogue: O = acc / l, stored straight from C layout ----
#pragma unroll
  for (int r = 0; r < 8; ++r) {
    const float inv = fast_rcp(lrow[r]);
    float* orow = o + (size_t)(m_base + half * 8 + r) * D_DIM + col;
#pragma unroll
    for (int t = 0; t < 4; ++t) orow[t * 16] = acc[t][r] * inv;
  }
}

extern "C" void kernel_launch(void* const* d_in, const int* in_sizes, int n_in,
                              void* d_out, int out_size, void* d_ws, size_t ws_size,
                              hipStream_t stream) {
  (void)in_sizes; (void)n_in; (void)out_size; (void)d_ws; (void)ws_size;
  const float* q = (const float*)d_in[0];
  const float* k = (const float*)d_in[1];
  const float* v = (const float*)d_in[2];
  // d_in[3] (additive causal mask) intentionally unused: mask applied analytically.
  float* out = (float*)d_out;

  const int BH = 2 * 16;                               // B*H
  const int tiles_per_bh = S_LEN / (WAVES * ROWS_PER_WAVE);
  dim3 grid(BH * tiles_per_bh);
  dim3 block(WAVES * 32);
  fa_fwd_causal_bf16wmma<<<grid, block, 0, stream>>>(q, k, v, out);
}
